// MultiHeadedSelfAttentionModule_39316130627952
// MI455X (gfx1250) — compile-verified, matrix operand order checked by reading, not run on hardware
//
#include <hip/hip_runtime.h>
#include <hip/hip_bf16.h>
#include <math.h>

#define DDIM 512
#define HNUM 8
#define DKD  64
#define BB   4
#define SS   1024
#define PPAD 2048   // padded (2S-1)=2047 -> 2048 rows per head, row 2047 == 0

typedef __attribute__((ext_vector_type(16))) __bf16 bf16x16;
typedef __attribute__((ext_vector_type(8)))  float  f32x8;

union FragU { bf16x16 v; uint4 q[2]; unsigned short u[16]; };

__device__ __forceinline__ unsigned short f2bf(float x) {
  unsigned u = __float_as_uint(x);
  u += 0x7fffu + ((u >> 16) & 1u);      // round-to-nearest-even
  return (unsigned short)(u >> 16);
}
__device__ __forceinline__ float bf2f(unsigned short h) {
  return __uint_as_float(((unsigned)h) << 16);
}

// NT fragment load (A or B operand) from row-major (row, K) bf16 memory.
// 16-bit WMMA operand layout: lane l -> row (l&15); K runs [kb, kb+8) and
// [kb+16, kb+24) where kb = kbase + (l>>4)*8.  Two 16-byte loads.
__device__ __forceinline__ bf16x16 load_frag(const unsigned short* base, int ld, int kbase) {
  const int lane = threadIdx.x & 31;
  const unsigned short* p = base + (size_t)(lane & 15) * ld + kbase + ((lane >> 4) << 3);
  FragU f;
  f.q[0] = *(const uint4*)(p);
  f.q[1] = *(const uint4*)(p + 16);
  return f.v;
}

__device__ __forceinline__ f32x8 wmma_bf(bf16x16 a, bf16x16 b, f32x8 c) {
  return __builtin_amdgcn_wmma_f32_16x16x32_bf16(false, a, false, b, (short)0, c, false, false);
}

// ---------------- prep kernels ----------------

__global__ void cvt_kernel(const float* __restrict__ src, unsigned short* __restrict__ dst, int n) {
  int i = blockIdx.x * blockDim.x + threadIdx.x;
  if (i < n) dst[i] = f2bf(src[i]);
}

// LayerNorm: one wave per row of (B*S, D)
__global__ void ln_kernel(const float* __restrict__ x, const float* __restrict__ g,
                          const float* __restrict__ be, unsigned short* __restrict__ out) {
  const int row = blockIdx.x;
  const int lane = threadIdx.x;
  const float* xr = x + (size_t)row * DDIM;
  float vals[16];
  float s = 0.f;
#pragma unroll
  for (int i = 0; i < 16; ++i) { float v = xr[lane + 32 * i]; vals[i] = v; s += v; }
#pragma unroll
  for (int o = 16; o; o >>= 1) s += __shfl_xor(s, o, 32);
  float mu = s * (1.0f / DDIM);
  float s2 = 0.f;
#pragma unroll
  for (int i = 0; i < 16; ++i) { float d = vals[i] - mu; s2 += d * d; }
#pragma unroll
  for (int o = 16; o; o >>= 1) s2 += __shfl_xor(s2, o, 32);
  float rstd = rsqrtf(s2 * (1.0f / DDIM) + 1e-5f);
#pragma unroll
  for (int i = 0; i < 16; ++i) {
    int c = lane + 32 * i;
    out[(size_t)row * DDIM + c] = f2bf((vals[i] - mu) * rstd * g[c] + be[c]);
  }
}

// Sinusoidal pos table, padded to 2048 rows (row 2047 = 0)
__global__ void pe_kernel(unsigned short* __restrict__ pe) {
  const int r = blockIdx.x;      // 0..2047
  const int t = threadIdx.x;     // 0..255 -> pair (2t, 2t+1)
  size_t o = (size_t)r * DDIM + 2 * t;
  if (r >= 2 * SS - 1) { pe[o] = 0; pe[o + 1] = 0; return; }
  float pos = (float)(r - (SS - 1));
  float dt = __expf((float)(2 * t) * (-9.210340371976184f / (float)DDIM)); // -ln(1e4)/D
  float a = pos * dt;
  pe[o]     = f2bf(sinf(a));
  pe[o + 1] = f2bf(cosf(a));
}

// ---------------- generic NT GEMM (one wave -> 16x64 tile) ----------------
#define MODE_QUV 0
#define MODE_K   1
#define MODE_V   2
#define MODE_P   3
#define MODE_O   4

__global__ void gemm_nt(const unsigned short* __restrict__ A, const unsigned short* __restrict__ Bw,
                        int M, int N, int K,
                        const float* __restrict__ bias, const float* __restrict__ pbu,
                        const float* __restrict__ pbv,
                        unsigned short* __restrict__ out0, unsigned short* __restrict__ out1,
                        float* __restrict__ outf, int mode) {
  const int lane = threadIdx.x;
  const int m0 = blockIdx.x * 16;
  const int n0 = blockIdx.y * 64;
  const unsigned short* arow = A + (size_t)m0 * K;
  const unsigned short* b0row = Bw + (size_t)(n0 +  0) * K;
  const unsigned short* b1row = Bw + (size_t)(n0 + 16) * K;
  const unsigned short* b2row = Bw + (size_t)(n0 + 32) * K;
  const unsigned short* b3row = Bw + (size_t)(n0 + 48) * K;

  f32x8 acc0 = (f32x8){0.f,0.f,0.f,0.f,0.f,0.f,0.f,0.f};
  f32x8 acc1 = acc0, acc2 = acc0, acc3 = acc0;

  for (int k = 0; k < K; k += 32) {
    // distinct fragment registers: all 10 b128 loads issue before one wait
    bf16x16 af  = load_frag(arow,  K, k);
    bf16x16 bf0 = load_frag(b0row, K, k);
    bf16x16 bf1 = load_frag(b1row, K, k);
    bf16x16 bf2 = load_frag(b2row, K, k);
    bf16x16 bf3 = load_frag(b3row, K, k);
    acc0 = wmma_bf(af, bf0, acc0);
    acc1 = wmma_bf(af, bf1, acc1);
    acc2 = wmma_bf(af, bf2, acc2);
    acc3 = wmma_bf(af, bf3, acc3);
  }

  f32x8 acc[4] = {acc0, acc1, acc2, acc3};
  const int m_l = 8 * (lane >> 4);
  const int n_l = lane & 15;
#pragma unroll
  for (int t = 0; t < 4; ++t) {
    const int n_g = n0 + t * 16 + n_l;
    const float badd = bias ? bias[n_g] : 0.0f;
#pragma unroll
    for (int v = 0; v < 8; ++v) {
      const int m_g = m0 + m_l + v;
      const float val = acc[t][v] + badd;
      if (mode == MODE_O) {
        outf[(size_t)m_g * N + n_g] = val;            // (B,S,D) flat
      } else if (mode == MODE_P) {
        const int hh = n_g >> 6, dk = n_g & 63;
        out0[((size_t)hh * PPAD + m_g) * DKD + dk] = f2bf(val);
      } else {
        const int bI = m_g >> 10, s = m_g & (SS - 1);
        const int hh = n_g >> 6, dk = n_g & 63;
        if (mode == MODE_V) {
          out0[(((size_t)bI * HNUM + hh) * DKD + dk) * SS + s] = f2bf(val); // V transposed
        } else {
          const size_t idx = (((size_t)bI * HNUM + hh) * SS + s) * DKD + dk;
          if (mode == MODE_QUV) {
            out0[idx] = f2bf(val + pbu[n_g]);
            out1[idx] = f2bf(val + pbv[n_g]);
          } else {
            out0[idx] = f2bf(val);                     // K
          }
        }
      }
    }
  }
}

// ---------------- fused attention: one wave per (b, h, 16-query tile) ----------------
// LDS: 16x1024 bf16 scores (32KB, reused in place for exp / attn weights)
//      + 16x32 f32 bd strip (2KB)  -> 34816 B/WG -> 9 workgroups per WGP
__global__ void attn_kernel(const unsigned short* __restrict__ quB, const unsigned short* __restrict__ qvB,
                            const unsigned short* __restrict__ kB,  const unsigned short* __restrict__ vtB,
                            const unsigned short* __restrict__ pB,  const unsigned char* __restrict__ mask,
                            unsigned short* __restrict__ aout) {
  extern __shared__ char smem[];
  unsigned short* sc = (unsigned short*)smem;              // 16*1024 bf16 (32KB)
  float* strip       = (float*)(smem + 16 * 1024 * 2);     // 16*32 f32    (2KB)

  const int lane = threadIdx.x;
  const int qt = blockIdx.x, h = blockIdx.y, b = blockIdx.z;
  const int q0 = qt * 16;
  const int bh = b * HNUM + h;
  const int m_l = 8 * (lane >> 4);
  const int n_l = lane & 15;

  const unsigned short* qub = quB + ((size_t)bh * SS + q0) * DKD;
  const unsigned short* qvb = qvB + ((size_t)bh * SS + q0) * DKD;
  const bf16x16 qu0 = load_frag(qub, DKD, 0);
  const bf16x16 qu1 = load_frag(qub, DKD, 32);
  const bf16x16 qv0 = load_frag(qvb, DKD, 0);
  const bf16x16 qv1 = load_frag(qvb, DKD, 32);

  for (int jt = 0; jt < SS / 16; ++jt) {
    // AC tile: q_u (16xDK) @ k_tile^T
    const unsigned short* kb = kB + ((size_t)bh * SS + jt * 16) * DKD;
    bf16x16 kf0 = load_frag(kb, DKD, 0);
    bf16x16 kf1 = load_frag(kb, DKD, 32);

    // BD strip: 16x32 over p rows [c0, c0+32); shifted[q,j] = bd[q, S+j-q]
    const int c0 = SS + jt * 16 - q0 - 15;   // >= 1
    const unsigned short* pb0 = pB + ((size_t)h * PPAD + c0) * DKD;
    const unsigned short* pb1 = pb0 + 16 * DKD;
    bf16x16 pf00 = load_frag(pb0, DKD, 0);
    bf16x16 pf01 = load_frag(pb0, DKD, 32);
    bf16x16 pf10 = load_frag(pb1, DKD, 0);
    bf16x16 pf11 = load_frag(pb1, DKD, 32);

    f32x8 ac = (f32x8){0.f,0.f,0.f,0.f,0.f,0.f,0.f,0.f};
    f32x8 bd0 = ac, bd1 = ac;
    ac  = wmma_bf(qu0, kf0,  ac);
    ac  = wmma_bf(qu1, kf1,  ac);
    bd0 = wmma_bf(qv0, pf00, bd0);
    bd0 = wmma_bf(qv1, pf01, bd0);
    bd1 = wmma_bf(qv0, pf10, bd1);
    bd1 = wmma_bf(qv1, pf11, bd1);

#pragma unroll
    for (int v = 0; v < 8; ++v) {
      strip[(m_l + v) * 32 + n_l]      = bd0[v];
      strip[(m_l + v) * 32 + 16 + n_l] = bd1[v];
    }
    __syncthreads();
#pragma unroll
    for (int v = 0; v < 8; ++v) {
      const int m = m_l + v;
      float s = (ac[v] + strip[m * 32 + (n_l + 15 - m)]) * 0.125f; // 1/sqrt(64)
      const int qg = q0 + m, jg = jt * 16 + n_l;
      if (mask[((size_t)b * SS + qg) * SS + jg]) s = -10000.0f;
      sc[m * SS + jg] = f2bf(s);
    }
    __syncthreads();
  }

  // wrap element of rel_shift: shifted[0, S-1] uses bd[1, 0] = q_v[1] . p[0]
  if (q0 == 0) {
    const unsigned short* qvr = qvB + ((size_t)bh * SS + 1) * DKD;
    const unsigned short* pr  = pB + (size_t)h * PPAD * DKD;
    float part = bf2f(qvr[2 * lane]) * bf2f(pr[2 * lane]) +
                 bf2f(qvr[2 * lane + 1]) * bf2f(pr[2 * lane + 1]);
#pragma unroll
    for (int o = 16; o; o >>= 1) part += __shfl_xor(part, o, 32);
    if (lane == 0 && !mask[((size_t)b * SS) * SS + (SS - 1)])
      sc[SS - 1] = f2bf(bf2f(sc[SS - 1]) + part * 0.125f);
  }
  __syncthreads();

  // softmax over each of the 16 rows (length S), bf16 in place
  for (int r = 0; r < 16; ++r) {
    float mx = -3.4e38f;
    for (int c = lane; c < SS; c += 32) mx = fmaxf(mx, bf2f(sc[r * SS + c]));
#pragma unroll
    for (int o = 16; o; o >>= 1) mx = fmaxf(mx, __shfl_xor(mx, o, 32));
    float sum = 0.f;
    for (int c = lane; c < SS; c += 32) {
      unsigned short eb = f2bf(__expf(bf2f(sc[r * SS + c]) - mx));
      sc[r * SS + c] = eb;
      sum += bf2f(eb);
    }
#pragma unroll
    for (int o = 16; o; o >>= 1) sum += __shfl_xor(sum, o, 32);
    const float rs = 1.0f / sum;
    for (int c = lane; c < SS; c += 32) sc[r * SS + c] = f2bf(bf2f(sc[r * SS + c]) * rs);
  }
  __syncthreads();

  // out = attn (16 x S) @ v (S x DK), v stored transposed (DK x S) -> NT
#pragma unroll
  for (int nt = 0; nt < 4; ++nt) {
    const unsigned short* vb = vtB + ((size_t)bh * DKD + nt * 16) * SS;
    f32x8 acc = (f32x8){0.f,0.f,0.f,0.f,0.f,0.f,0.f,0.f};
    for (int kt = 0; kt < SS; kt += 32) {
      bf16x16 af = load_frag(sc, SS, kt);
      bf16x16 bf = load_frag(vb, SS, kt);
      acc = wmma_bf(af, bf, acc);
    }
#pragma unroll
    for (int v = 0; v < 8; ++v) {
      const int m = m_l + v;
      aout[((size_t)b * SS + q0 + m) * DDIM + h * DKD + nt * 16 + n_l] = f2bf(acc[v]);
    }
  }
}

// ---------------- host orchestration ----------------
extern "C" void kernel_launch(void* const* d_in, const int* in_sizes, int n_in,
                              void* d_out, int out_size, void* d_ws, size_t ws_size,
                              hipStream_t stream) {
  (void)in_sizes; (void)n_in; (void)out_size; (void)ws_size;
  const float* x   = (const float*)d_in[0];
  const unsigned char* mask = (const unsigned char*)d_in[1];
  const float* Wq  = (const float*)d_in[2];
  const float* bq  = (const float*)d_in[3];
  const float* Wk  = (const float*)d_in[4];
  const float* bk  = (const float*)d_in[5];
  const float* Wv  = (const float*)d_in[6];
  const float* bv  = (const float*)d_in[7];
  const float* Wo  = (const float*)d_in[8];
  const float* bo  = (const float*)d_in[9];
  const float* Wp  = (const float*)d_in[10];
  const float* pbu = (const float*)d_in[11];
  const float* pbv = (const float*)d_in[12];
  const float* lg  = (const float*)d_in[13];
  const float* lb  = (const float*)d_in[14];
  float* out = (float*)d_out;

  char* ws = (char*)d_ws;
  unsigned short* xln = (unsigned short*)(ws + 0);          // 4096x512 bf16   (4MB)
  unsigned short* pe  = (unsigned short*)(ws + 4194304);    // 2048x512 bf16   (2MB)
  unsigned short* wqb = (unsigned short*)(ws + 6291456);    // 5 x 512x512 bf16
  unsigned short* wkb = wqb + 262144;
  unsigned short* wvb = wkb + 262144;
  unsigned short* wob = wvb + 262144;
  unsigned short* wpb = wob + 262144;
  unsigned short* qu  = (unsigned short*)(ws + 8912896);    // (B,H,S,DK) bf16 (4MB)
  unsigned short* qv  = (unsigned short*)(ws + 13107200);   // (B,H,S,DK)      (4MB)
  unsigned short* kkb = (unsigned short*)(ws + 17301504);   // (B,H,S,DK)      (4MB)
  unsigned short* vtb = (unsigned short*)(ws + 21495808);   // (B,H,DK,S)      (4MB)
  unsigned short* pmt = (unsigned short*)(ws + 25690112);   // (H, 2048+pad, DK) (~2MB)
  unsigned short* ao  = (unsigned short*)(ws + 27789312);   // (B*S, D) bf16   (4MB)

  // weight conversions f32 -> bf16
  cvt_kernel<<<1024, 256, 0, stream>>>(Wq, wqb, 262144);
  cvt_kernel<<<1024, 256, 0, stream>>>(Wk, wkb, 262144);
  cvt_kernel<<<1024, 256, 0, stream>>>(Wv, wvb, 262144);
  cvt_kernel<<<1024, 256, 0, stream>>>(Wo, wob, 262144);
  cvt_kernel<<<1024, 256, 0, stream>>>(Wp, wpb, 262144);

  ln_kernel<<<BB * SS, 32, 0, stream>>>(x, lg, lb, xln);
  pe_kernel<<<PPAD, 256, 0, stream>>>(pe);

  // Q (fused +pos_bias_u / +pos_bias_v), K, V(transposed), P projections
  gemm_nt<<<dim3(256, 8), 32, 0, stream>>>(xln, wqb, 4096, 512, 512, bq, pbu, pbv, qu, qv, nullptr, MODE_QUV);
  gemm_nt<<<dim3(256, 8), 32, 0, stream>>>(xln, wkb, 4096, 512, 512, bk, nullptr, nullptr, kkb, nullptr, nullptr, MODE_K);
  gemm_nt<<<dim3(256, 8), 32, 0, stream>>>(xln, wvb, 4096, 512, 512, bv, nullptr, nullptr, vtb, nullptr, nullptr, MODE_V);
  gemm_nt<<<dim3(128, 8), 32, 0, stream>>>(pe, wpb, 2048, 512, 512, nullptr, nullptr, nullptr, pmt, nullptr, nullptr, MODE_P);

  // fused scores + rel-shift + softmax + attn@V  (dynamic LDS: 32KB + 2KB)
  attn_kernel<<<dim3(SS / 16, HNUM, BB), 32, 34816, stream>>>(qu, qv, kkb, vtb, pmt, mask, ao);

  // output projection -> f32 d_out
  gemm_nt<<<dim3(256, 8), 32, 0, stream>>>(ao, wob, 4096, 512, 512, bo, nullptr, nullptr, nullptr, nullptr, out, MODE_O);
}
